// SpatioTemporalPerceiverResampler_31001073942624
// MI455X (gfx1250) — matrix operand order repaired and checked by reading, not compiled, not run on hardware
//
#include <hip/hip_runtime.h>
#include <hip/hip_bf16.h>
#include <stdint.h>

// ---------------------------------------------------------------------------
// SpatioTemporalPerceiverResampler for MI455X (gfx1250, wave32, WMMA)
// All heavy GEMMs + attention use v_wmma_f32_16x16x32_bf16.
// Fragments are fed by ds_load_b128 (weights pre-packed to [K/32][N][32]).
// ---------------------------------------------------------------------------

typedef __attribute__((ext_vector_type(16))) __bf16 v16bf;
typedef __attribute__((ext_vector_type(8)))  float  v8f;

#define D_MODEL 512
#define NHEAD   8
#define DHEAD   64
#define NBLK    4
#define DMLP    2048
#define NLAT    32
#define BB      8
#define TT      128
#define BT      (BB*TT)        // 1024
#define MROWS   (BT*NLAT)      // 32768

union AF { v16bf v; __bf16 e[16]; uint4 q[2]; };
union CF { v8f  v; float  e[8];  };

// ------------------------------ GEMM ---------------------------------------
// C[M,N] = A[M,K](bf16, row-major) @ Bp (bf16 packed [K/32][N][32])
// (+bias f32[N]) (+resid f32[M,N]); out f32 (outF) or bf16 (outH).
// M,N multiples of 128; K multiple of 32.
__global__ __launch_bounds__(256)
void gemm_bf16_wmma(const __bf16* __restrict__ A, const __bf16* __restrict__ Bp,
                    int M, int N, int K,
                    const float* __restrict__ bias,
                    const float* __restrict__ resid,
                    float* __restrict__ outF, __bf16* __restrict__ outH)
{
    __shared__ __bf16 As[128][32];
    __shared__ __bf16 Bs[128][32];   // [n_local][kk]
    const int bm = blockIdx.y * 128;
    const int bn = blockIdx.x * 128;
    const int tid  = threadIdx.x;
    const int wave = tid >> 5, lane = tid & 31;
    const int l15  = lane & 15, hi = lane >> 4;
    const int wr = wave >> 1, wc = wave & 1;   // 4x2 wave grid -> 32x64 per wave

    CF acc[2][4];
    #pragma unroll
    for (int i = 0; i < 2; i++)
        #pragma unroll
        for (int j = 0; j < 4; j++)
            #pragma unroll
            for (int r = 0; r < 8; r++) acc[i][j].e[r] = 0.f;

    for (int k0 = 0; k0 < K; k0 += 32) {
        // stage A tile 128x32 (512 x 16B chunks, coalesced)
        #pragma unroll
        for (int it = 0; it < 2; it++) {
            int off = (tid + it * 256) * 8;
            int r = off >> 5, c = off & 31;
            *(uint4*)(&As[r][c]) =
                *(const uint4*)(A + (size_t)(bm + r) * K + k0 + c);
        }
        // stage B tile: contiguous 8KB slab of the packed weight
        const __bf16* slab = Bp + ((size_t)(k0 >> 5) * N + bn) * 32;
        #pragma unroll
        for (int it = 0; it < 2; it++) {
            int off = (tid + it * 256) * 8;
            *(uint4*)(&Bs[off >> 5][off & 31]) = *(const uint4*)(slab + off);
        }
        if (k0 + 32 < K)
            __builtin_prefetch(A + (size_t)(bm + (tid >> 1)) * K + k0 + 32, 0, 3);
        __syncthreads();

        AF af[2], bfr[4];
        #pragma unroll
        for (int i = 0; i < 2; i++) {
            int arow = (wr * 2 + i) * 16 + l15;
            af[i].q[0] = *(const uint4*)(&As[arow][8 * hi]);
            af[i].q[1] = *(const uint4*)(&As[arow][16 + 8 * hi]);
        }
        #pragma unroll
        for (int j4 = 0; j4 < 4; j4++) {
            int bcol = (wc * 4 + j4) * 16 + l15;
            bfr[j4].q[0] = *(const uint4*)(&Bs[bcol][8 * hi]);
            bfr[j4].q[1] = *(const uint4*)(&Bs[bcol][16 + 8 * hi]);
        }
        #pragma unroll
        for (int i = 0; i < 2; i++)
            #pragma unroll
            for (int j4 = 0; j4 < 4; j4++)
                acc[i][j4].v = __builtin_amdgcn_wmma_f32_16x16x32_bf16(
                    false, af[i].v, false, bfr[j4].v,
                    (short)0, acc[i][j4].v, false, false);
        __syncthreads();
    }

    #pragma unroll
    for (int i = 0; i < 2; i++)
        #pragma unroll
        for (int j4 = 0; j4 < 4; j4++)
            #pragma unroll
            for (int r = 0; r < 8; r++) {
                int grow = bm + (wr * 2 + i) * 16 + r + 8 * hi;
                int gcol = bn + (wc * 4 + j4) * 16 + l15;
                float v = acc[i][j4].e[r];
                if (bias)  v += bias[gcol];
                if (resid) v += resid[(size_t)grow * N + gcol];
                if (outF)  outF[(size_t)grow * N + gcol] = v;
                else       outH[(size_t)grow * N + gcol] = (__bf16)v;
            }
}

// --------------------------- Attention -------------------------------------
// grid.x = batches*NHEAD. Q:[Nb,Lq,512] K,V:[Nb,Lk,512] bf16, head h uses cols
// h*64..h*64+63. O:[Nb,Lq,512] bf16. Lq multiple of 16, Lk multiple of 32 (<=128).
// softmax(q k^T / sqrt(dh)) v ; the 1/8 = dh^-0.5 is applied to the scores.
__global__ __launch_bounds__(256)
void attn_wmma(const __bf16* __restrict__ Q, const __bf16* __restrict__ Kv,
               const __bf16* __restrict__ V, __bf16* __restrict__ O,
               int Lq, int Lk)
{
    __shared__ __bf16 Qs[128][DHEAD];
    __shared__ __bf16 Ks[128][DHEAD];
    __shared__ __bf16 Vs[128][DHEAD];
    __shared__ __bf16 S [128][128];
    const int n = blockIdx.x >> 3;
    const int h = blockIdx.x & 7;
    const int tid = threadIdx.x;
    const __bf16* qg = Q  + ((size_t)n * Lq) * D_MODEL + h * DHEAD;
    const __bf16* kg = Kv + ((size_t)n * Lk) * D_MODEL + h * DHEAD;
    const __bf16* vg = V  + ((size_t)n * Lk) * D_MODEL + h * DHEAD;

    for (int i = tid; i < Lq * 8; i += 256) {      // 8 x uint4 per 64-wide row
        int r = i >> 3, c8 = (i & 7) * 8;
        *(uint4*)(&Qs[r][c8]) = *(const uint4*)(qg + (size_t)r * D_MODEL + c8);
    }
    for (int i = tid; i < Lk * 8; i += 256) {
        int r = i >> 3, c8 = (i & 7) * 8;
        *(uint4*)(&Ks[r][c8]) = *(const uint4*)(kg + (size_t)r * D_MODEL + c8);
        *(uint4*)(&Vs[r][c8]) = *(const uint4*)(vg + (size_t)r * D_MODEL + c8);
    }
    __syncthreads();

    const int wave = tid >> 5, lane = tid & 31, l15 = lane & 15, hi = lane >> 4;
    const int nqt = Lq >> 4, nkt = Lk >> 4;

    // S = (Q K^T) * dh^-0.5
    for (int qt = wave; qt < nqt; qt += 8) {
        int qrow = qt * 16 + l15;
        for (int kt = 0; kt < nkt; kt++) {
            int krow = kt * 16 + l15;
            CF acc;
            #pragma unroll
            for (int r = 0; r < 8; r++) acc.e[r] = 0.f;
            #pragma unroll
            for (int kk = 0; kk < DHEAD; kk += 32) {
                AF a, b;
                a.q[0] = *(const uint4*)(&Qs[qrow][kk + 8 * hi]);
                a.q[1] = *(const uint4*)(&Qs[qrow][kk + 16 + 8 * hi]);
                b.q[0] = *(const uint4*)(&Ks[krow][kk + 8 * hi]);
                b.q[1] = *(const uint4*)(&Ks[krow][kk + 16 + 8 * hi]);
                acc.v = __builtin_amdgcn_wmma_f32_16x16x32_bf16(
                    false, a.v, false, b.v, (short)0, acc.v, false, false);
            }
            #pragma unroll
            for (int r = 0; r < 8; r++)
                S[qt * 16 + r + 8 * hi][kt * 16 + l15] = (__bf16)(acc.e[r] * 0.125f);
        }
    }
    __syncthreads();

    // row softmax
    for (int r = tid; r < Lq; r += 256) {
        float mx = -1e30f;
        for (int c = 0; c < Lk; c++) { float v = (float)S[r][c]; mx = v > mx ? v : mx; }
        float sum = 0.f;
        for (int c = 0; c < Lk; c++) {
            float e = __expf((float)S[r][c] - mx); sum += e; S[r][c] = (__bf16)e;
        }
        float inv = 1.f / sum;
        for (int c = 0; c < Lk; c++) S[r][c] = (__bf16)((float)S[r][c] * inv);
    }
    __syncthreads();

    // O = P @ V
    for (int qt = wave; qt < nqt; qt += 8) {
        int qrow = qt * 16 + l15;
        CF oacc[4];
        #pragma unroll
        for (int ct = 0; ct < 4; ct++)
            #pragma unroll
            for (int r = 0; r < 8; r++) oacc[ct].e[r] = 0.f;
        for (int ks = 0; ks < Lk; ks += 32) {
            AF a;
            a.q[0] = *(const uint4*)(&S[qrow][ks + 8 * hi]);
            a.q[1] = *(const uint4*)(&S[qrow][ks + 16 + 8 * hi]);
            #pragma unroll
            for (int ct = 0; ct < 4; ct++) {
                AF b;   // V column slice (transpose gather)
                #pragma unroll
                for (int j = 0; j < 16; j++)
                    b.e[j] = Vs[ks + 8 * hi + (j & 7) + ((j >> 3) << 4)][ct * 16 + l15];
                oacc[ct].v = __builtin_amdgcn_wmma_f32_16x16x32_bf16(
                    false, a.v, false, b.v, (short)0, oacc[ct].v, false, false);
            }
        }
        #pragma unroll
        for (int ct = 0; ct < 4; ct++)
            #pragma unroll
            for (int r = 0; r < 8; r++) {
                int row = qt * 16 + r + 8 * hi;
                O[((size_t)n * Lq + row) * D_MODEL + h * DHEAD + ct * 16 + l15] =
                    (__bf16)oacc[ct].e[r];
            }
    }
}

// --------------------------- Elementwise -----------------------------------
// RMS-norm one 512-wide row per block; write bf16 with row remapping so the
// kv concat [s_n ; l_n] is produced in place.
__global__ __launch_bounds__(128)
void rms_cast(const float* __restrict__ in, const float* __restrict__ gain,
              __bf16* __restrict__ out, int rpb_in, int rpb_out, int off_out)
{
    int r = blockIdx.x;
    const float* x = in + (size_t)r * D_MODEL;
    __shared__ float red[128];
    float s = 0.f;
    for (int c = threadIdx.x; c < D_MODEL; c += 128) { float v = x[c]; s += v * v; }
    red[threadIdx.x] = s; __syncthreads();
    for (int o = 64; o > 0; o >>= 1) {
        if (threadIdx.x < o) red[threadIdx.x] += red[threadIdx.x + o];
        __syncthreads();
    }
    float sc = rsqrtf(red[0] / (float)D_MODEL + 1e-6f);
    size_t orow = (size_t)(r / rpb_in) * rpb_out + (r % rpb_in) + off_out;
    __bf16* o = out + orow * D_MODEL;
    for (int c = threadIdx.x; c < D_MODEL; c += 128)
        o[c] = (__bf16)(x[c] * sc * gain[c]);
}

// f32 [NB,K,N] weight -> bf16 packed [NB][K/32][N][32]
__global__ void pack_weight_bf16(const float* __restrict__ in,
                                 __bf16* __restrict__ out,
                                 int K, int N, size_t total)
{
    size_t idx = (size_t)blockIdx.x * 256 + threadIdx.x;
    if (idx >= total) return;
    size_t kn = (size_t)K * N;
    size_t b = idx / kn;
    size_t rem = idx - b * kn;
    int k = (int)(rem / N), nn = (int)(rem % N);
    out[b * kn + (((size_t)(k >> 5)) * N + nn) * 32 + (k & 31)] = (__bf16)in[idx];
}

__global__ void bcast_lat(const float* __restrict__ ql, float* __restrict__ lat,
                          size_t total)
{
    size_t i = (size_t)blockIdx.x * 256 + threadIdx.x;
    if (i < total) lat[i] = ql[i & (NLAT * D_MODEL - 1)];
}

__global__ void copy_f32(const float* __restrict__ in, float* __restrict__ out,
                         size_t n)
{
    size_t i = (size_t)blockIdx.x * 256 + threadIdx.x;
    if (i < n) out[i] = in[i];
}

// RoPE in place on bf16 Q and K (first 32 dims of each head). row%T = position.
__global__ void rope_bf16(__bf16* __restrict__ q, __bf16* __restrict__ k,
                          int totalRows)
{
    int idx = blockIdx.x * 256 + threadIdx.x;
    int total = totalRows * NHEAD * 16;
    if (idx >= total) return;
    int i = idx & 15;
    int h = (idx >> 4) & 7;
    int r = idx >> 7;
    int t = r & (TT - 1);
    float freq = __powf(10000.f, -(float)(2 * i) / 32.f);
    float sn, cs;
    __sincosf((float)t * freq, &sn, &cs);
    size_t base = (size_t)r * D_MODEL + h * DHEAD + 2 * i;
    float x1 = (float)q[base], x2 = (float)q[base + 1];
    q[base]     = (__bf16)(x1 * cs - x2 * sn);
    q[base + 1] = (__bf16)(x1 * sn + x2 * cs);
    x1 = (float)k[base]; x2 = (float)k[base + 1];
    k[base]     = (__bf16)(x1 * cs - x2 * sn);
    k[base + 1] = (__bf16)(x1 * sn + x2 * cs);
}

// h:[rows,4096] -> out:[rows,2048] = a * gelu_tanh(g)
__global__ void geglu_bf16(const __bf16* __restrict__ h,
                           __bf16* __restrict__ out, size_t total)
{
    size_t idx = (size_t)blockIdx.x * 256 + threadIdx.x;
    if (idx >= total) return;
    size_t r = idx / DMLP, c = idx % DMLP;
    float a = (float)h[r * (2 * DMLP) + c];
    float g = (float)h[r * (2 * DMLP) + DMLP + c];
    float gel = 0.5f * g * (1.f + tanhf(0.7978845608028654f * (g + 0.044715f * g * g * g)));
    out[idx] = (__bf16)(a * gel);
}

// lat [B,T,L,D] -> x [B,L,T,D]
__global__ void perm_lat_to_x(const float* __restrict__ in, float* __restrict__ out)
{
    size_t idx = (size_t)blockIdx.x * 256 + threadIdx.x;
    if (idx >= (size_t)MROWS * D_MODEL) return;
    int d = idx & 511; size_t rest = idx >> 9;
    int l = rest & 31; rest >>= 5;
    int t = rest & 127; int b = (int)(rest >> 7);
    out[((((size_t)b * NLAT + l) * TT + t) << 9) + d] = in[idx];
}
// x [B,L,T,D] -> lat [B,T,L,D]
__global__ void perm_x_to_lat(const float* __restrict__ in, float* __restrict__ out)
{
    size_t idx = (size_t)blockIdx.x * 256 + threadIdx.x;
    if (idx >= (size_t)MROWS * D_MODEL) return;
    int d = idx & 511; size_t rest = idx >> 9;
    int t = rest & 127; rest >>= 7;
    int l = rest & 31; int b = (int)(rest >> 5);
    out[((((size_t)b * TT + t) * NLAT + l) << 9) + d] = in[idx];
}

// ---------------------------------------------------------------------------
extern "C" void kernel_launch(void* const* d_in, const int* in_sizes, int n_in,
                              void* d_out, int out_size, void* d_ws, size_t ws_size,
                              hipStream_t stream)
{
    (void)in_sizes; (void)n_in; (void)out_size; (void)ws_size;
    const float* src   = (const float*)d_in[0];
    const float* qlat  = (const float*)d_in[1];
    const float* ca_ln = (const float*)d_in[2];
    const float* ca_wq = (const float*)d_in[3];
    const float* ca_wk = (const float*)d_in[4];
    const float* ca_bk = (const float*)d_in[5];
    const float* ca_wv = (const float*)d_in[6];
    const float* ca_wo = (const float*)d_in[7];
    const float* ca_w1 = (const float*)d_in[8];
    const float* ca_w2 = (const float*)d_in[9];
    const float* ta_ln = (const float*)d_in[10];
    const float* ta_wq = (const float*)d_in[11];
    const float* ta_wk = (const float*)d_in[12];
    const float* ta_bk = (const float*)d_in[13];
    const float* ta_wv = (const float*)d_in[14];
    const float* ta_wo = (const float*)d_in[15];
    const float* ta_w1 = (const float*)d_in[16];
    const float* ta_w2 = (const float*)d_in[17];

    char* p = (char*)d_ws;
    auto alloc = [&](size_t bytes) -> void* {
        void* r = (void*)p; p += (bytes + 255) & ~(size_t)255; return r;
    };
    const size_t NE  = (size_t)MROWS * D_MODEL;    // 16,777,216
    const size_t NKV = (size_t)BT * 96 * D_MODEL;  // 50,331,648
    const int    CH  = 4096;                       // MLP row chunk

    float*  lat  = (float*) alloc(NE * 4);
    float*  xbuf = (float*) alloc(NE * 4);
    __bf16* lnb  = (__bf16*)alloc(NE * 2);
    __bf16* kvb  = (__bf16*)alloc(NKV * 2);
    __bf16* qb   = (__bf16*)alloc(NE * 2);
    __bf16* kb   = (__bf16*)alloc(NKV * 2);
    __bf16* vb   = (__bf16*)alloc(NKV * 2);
    __bf16* ob   = (__bf16*)alloc(NE * 2);
    __bf16* hid  = (__bf16*)alloc((size_t)CH * 2 * DMLP * 2);
    __bf16* ggb  = (__bf16*)alloc((size_t)CH * DMLP * 2);

    const size_t WPROJ = (size_t)NBLK * D_MODEL * D_MODEL;
    const size_t W1SZ  = (size_t)NBLK * D_MODEL * 2 * DMLP;
    const size_t W2SZ  = (size_t)NBLK * DMLP * D_MODEL;
    __bf16* ca_wq_h = (__bf16*)alloc(WPROJ * 2);
    __bf16* ca_wk_h = (__bf16*)alloc(WPROJ * 2);
    __bf16* ca_wv_h = (__bf16*)alloc(WPROJ * 2);
    __bf16* ca_wo_h = (__bf16*)alloc(WPROJ * 2);
    __bf16* ca_w1_h = (__bf16*)alloc(W1SZ * 2);
    __bf16* ca_w2_h = (__bf16*)alloc(W2SZ * 2);
    __bf16* ta_wq_h = (__bf16*)alloc(WPROJ * 2);
    __bf16* ta_wk_h = (__bf16*)alloc(WPROJ * 2);
    __bf16* ta_wv_h = (__bf16*)alloc(WPROJ * 2);
    __bf16* ta_wo_h = (__bf16*)alloc(WPROJ * 2);
    __bf16* ta_w1_h = (__bf16*)alloc(W1SZ * 2);
    __bf16* ta_w2_h = (__bf16*)alloc(W2SZ * 2);

    auto packW = [&](const float* s, __bf16* d, int K, int N, size_t total) {
        unsigned g = (unsigned)((total + 255) / 256);
        pack_weight_bf16<<<g, 256, 0, stream>>>(s, d, K, N, total);
    };
    packW(ca_wq, ca_wq_h, D_MODEL, D_MODEL, WPROJ);
    packW(ca_wk, ca_wk_h, D_MODEL, D_MODEL, WPROJ);
    packW(ca_wv, ca_wv_h, D_MODEL, D_MODEL, WPROJ);
    packW(ca_wo, ca_wo_h, D_MODEL, D_MODEL, WPROJ);
    packW(ca_w1, ca_w1_h, D_MODEL, 2 * DMLP, W1SZ);
    packW(ca_w2, ca_w2_h, DMLP, D_MODEL, W2SZ);
    packW(ta_wq, ta_wq_h, D_MODEL, D_MODEL, WPROJ);
    packW(ta_wk, ta_wk_h, D_MODEL, D_MODEL, WPROJ);
    packW(ta_wv, ta_wv_h, D_MODEL, D_MODEL, WPROJ);
    packW(ta_wo, ta_wo_h, D_MODEL, D_MODEL, WPROJ);
    packW(ta_w1, ta_w1_h, D_MODEL, 2 * DMLP, W1SZ);
    packW(ta_w2, ta_w2_h, DMLP, D_MODEL, W2SZ);

    auto gemm = [&](const __bf16* A, const __bf16* Bm, int M, int N, int K,
                    const float* bias, const float* resid,
                    float* oF, __bf16* oH) {
        dim3 g((unsigned)(N / 128), (unsigned)(M / 128), 1);
        gemm_bf16_wmma<<<g, 256, 0, stream>>>(A, Bm, M, N, K, bias, resid, oF, oH);
    };

    // latents = broadcast(query_latents)
    bcast_lat<<<(unsigned)((NE + 255) / 256), 256, 0, stream>>>(qlat, lat, NE);

    const unsigned PGRID = (unsigned)(NE / 256);

    for (int b = 0; b < NBLK; b++) {
        const float* lnq = ca_ln + ((size_t)b * 3 + 0) * D_MODEL;
        const float* lns = ca_ln + ((size_t)b * 3 + 1) * D_MODEL;
        const float* lnm = ca_ln + ((size_t)b * 3 + 2) * D_MODEL;

        // ---- cross attention ----
        rms_cast<<<MROWS, 128, 0, stream>>>(lat, lnq, lnb, 32, 32, 0);
        rms_cast<<<MROWS, 128, 0, stream>>>(lat, lnq, kvb, 32, 96, 64);
        rms_cast<<<BT * 64, 128, 0, stream>>>(src, lns, kvb, 64, 96, 0);

        gemm(lnb, ca_wq_h + (size_t)b * D_MODEL * D_MODEL, MROWS, D_MODEL, D_MODEL,
             nullptr, nullptr, nullptr, qb);
        gemm(kvb, ca_wk_h + (size_t)b * D_MODEL * D_MODEL, BT * 96, D_MODEL, D_MODEL,
             ca_bk + (size_t)b * D_MODEL, nullptr, nullptr, kb);
        gemm(kvb, ca_wv_h + (size_t)b * D_MODEL * D_MODEL, BT * 96, D_MODEL, D_MODEL,
             nullptr, nullptr, nullptr, vb);

        attn_wmma<<<BT * NHEAD, 256, 0, stream>>>(qb, kb, vb, ob, NLAT, 96);

        gemm(ob, ca_wo_h + (size_t)b * D_MODEL * D_MODEL, MROWS, D_MODEL, D_MODEL,
             nullptr, lat, lat, nullptr);

        // ---- cross MLP (chunked) ----
        rms_cast<<<MROWS, 128, 0, stream>>>(lat, lnm, lnb, 32, 32, 0);
        for (int m0 = 0; m0 < MROWS; m0 += CH) {
            gemm(lnb + (size_t)m0 * D_MODEL, ca_w1_h + (size_t)b * D_MODEL * 2 * DMLP,
                 CH, 2 * DMLP, D_MODEL, nullptr, nullptr, nullptr, hid);
            geglu_bf16<<<(unsigned)(((size_t)CH * DMLP + 255) / 256), 256, 0, stream>>>(
                hid, ggb, (size_t)CH * DMLP);
            gemm(ggb, ca_w2_h + (size_t)b * DMLP * D_MODEL, CH, D_MODEL, DMLP,
                 nullptr, lat + (size_t)m0 * D_MODEL, lat + (size_t)m0 * D_MODEL, nullptr);
        }

        // ---- temporal attention ----
        perm_lat_to_x<<<PGRID, 256, 0, stream>>>(lat, xbuf);
        const float* tln0 = ta_ln + ((size_t)b * 2 + 0) * D_MODEL;
        const float* tln1 = ta_ln + ((size_t)b * 2 + 1) * D_MODEL;
        rms_cast<<<MROWS, 128, 0, stream>>>(xbuf, tln0, lnb, 32, 32, 0);

        gemm(lnb, ta_wq_h + (size_t)b * D_MODEL * D_MODEL, MROWS, D_MODEL, D_MODEL,
             nullptr, nullptr, nullptr, qb);
        gemm(lnb, ta_wk_h + (size_t)b * D_MODEL * D_MODEL, MROWS, D_MODEL, D_MODEL,
             ta_bk + (size_t)b * D_MODEL, nullptr, nullptr, kb);
        gemm(lnb, ta_wv_h + (size_t)b * D_MODEL * D_MODEL, MROWS, D_MODEL, D_MODEL,
             nullptr, nullptr, nullptr, vb);

        rope_bf16<<<(unsigned)((MROWS * NHEAD * 16 + 255) / 256), 256, 0, stream>>>(
            qb, kb, MROWS);

        attn_wmma<<<BB * NLAT * NHEAD, 256, 0, stream>>>(qb, kb, vb, ob, TT, TT);

        gemm(ob, ta_wo_h + (size_t)b * D_MODEL * D_MODEL, MROWS, D_MODEL, D_MODEL,
             nullptr, xbuf, xbuf, nullptr);

        // ---- temporal MLP (chunked) ----
        rms_cast<<<MROWS, 128, 0, stream>>>(xbuf, tln1, lnb, 32, 32, 0);
        for (int m0 = 0; m0 < MROWS; m0 += CH) {
            gemm(lnb + (size_t)m0 * D_MODEL, ta_w1_h + (size_t)b * D_MODEL * 2 * DMLP,
                 CH, 2 * DMLP, D_MODEL, nullptr, nullptr, nullptr, hid);
            geglu_bf16<<<(unsigned)(((size_t)CH * DMLP + 255) / 256), 256, 0, stream>>>(
                hid, ggb, (size_t)CH * DMLP);
            gemm(ggb, ta_w2_h + (size_t)b * DMLP * D_MODEL, CH, D_MODEL, DMLP,
                 nullptr, xbuf + (size_t)m0 * D_MODEL, xbuf + (size_t)m0 * D_MODEL, nullptr);
        }

        perm_x_to_lat<<<PGRID, 256, 0, stream>>>(xbuf, lat);
    }

    // latents.reshape(B, T, L*D) is exactly the flat [BT, L, D] layout.
    copy_f32<<<(unsigned)((NE + 255) / 256), 256, 0, stream>>>(lat, (float*)d_out, NE);
}
